// Attention_53927609368925
// MI455X (gfx1250) — compile-verified, matrix operand order checked by reading, not run on hardware
//
#include <hip/hip_runtime.h>

typedef __attribute__((ext_vector_type(16))) _Float16 v16h;
typedef __attribute__((ext_vector_type(8)))  _Float16 v8h;
typedef __attribute__((ext_vector_type(8)))  float    v8f;

#define DMODEL 768
#define HD     64
#define NH     12
#define BB     2
#define SS     4096
#define MM     (BB * SS)          // 8192 rows of the token matrix

// ---------------------------------------------------------------------------
// WMMA helpers
// ---------------------------------------------------------------------------
__device__ __forceinline__ v8f wmma16(v16h a, v16h b, v8f c) {
  // D = A(16x32 f16) * B(32x16 f16) + C(16x16 f32)
  return __builtin_amdgcn_wmma_f32_16x16x32_f16(false, a, false, b,
                                                (short)0, c, false, false);
}

// 8 contiguous halfs at p, 8 more at p+16 -> one A/B fragment register set.
__device__ __forceinline__ v16h ldfrag(const _Float16* p) {
  v8h lo = *(const v8h*)(p);
  v8h hi = *(const v8h*)(p + 16);
  v16h r;
#pragma unroll
  for (int i = 0; i < 8; ++i) { r[i] = lo[i]; r[i + 8] = hi[i]; }
  return r;
}

// Fragment from a row-major f16 matrix (contraction along contiguous axis).
__device__ __forceinline__ v16h ldfrag_g(const _Float16* __restrict__ base,
                                         int ld, int row, int kbase) {
  int lane = threadIdx.x & 31;
  int sel  = (lane < 16) ? 0 : 8;
  return ldfrag(base + (size_t)row * ld + kbase + sel);
}

// 16-lane (half-wave) reductions; xor masks 1..8 never cross the 16-group.
__device__ __forceinline__ float rmax16(float v) {
#pragma unroll
  for (int s = 1; s < 16; s <<= 1) v = fmaxf(v, __shfl_xor(v, s, 32));
  return v;
}
__device__ __forceinline__ float rsum16(float v) {
#pragma unroll
  for (int s = 1; s < 16; s <<= 1) v += __shfl_xor(v, s, 32);
  return v;
}

// ---------------------------------------------------------------------------
// Stage 0: f32 -> f16 conversion
// ---------------------------------------------------------------------------
__global__ void cvt_f32_f16(const float* __restrict__ in,
                            _Float16* __restrict__ out, int n) {
  int i = blockIdx.x * blockDim.x + threadIdx.x;
  int stride = gridDim.x * blockDim.x;
  for (; i < n; i += stride) out[i] = (_Float16)in[i];
}

// ---------------------------------------------------------------------------
// Double-buffered 32(M) x 64(N) x 768(K) tile GEMM core: y = X @ W^T.
// c[0..3] = rows mbase..mbase+15 (N-tiles 0..3), c[4..7] = rows +16..+31.
// ---------------------------------------------------------------------------
__device__ __forceinline__ void load_step(const _Float16* __restrict__ X,
                                          const _Float16* __restrict__ W,
                                          int mbase, int nbase, int kb, int col,
                                          v16h& aLo, v16h& aHi, v16h (&b)[4]) {
  aLo = ldfrag_g(X, DMODEL, mbase + col, kb);
  aHi = ldfrag_g(X, DMODEL, mbase + 16 + col, kb);
#pragma unroll
  for (int t = 0; t < 4; ++t)
    b[t] = ldfrag_g(W, DMODEL, nbase + 16 * t + col, kb);
}

__device__ __forceinline__ void gemm_tile(const _Float16* __restrict__ X,
                                          const _Float16* __restrict__ W,
                                          int mbase, int nbase, int col,
                                          v8f (&c)[8]) {
  v16h a0L, a0H, b0[4];   // even k-step buffer
  v16h a1L, a1H, b1[4];   // odd  k-step buffer
  load_step(X, W, mbase, nbase, 0, col, a0L, a0H, b0);
  for (int kb = 0; kb < DMODEL; kb += 64) {
    // issue odd-step loads, then consume even step (loads overlap WMMA)
    load_step(X, W, mbase, nbase, kb + 32, col, a1L, a1H, b1);
#pragma unroll
    for (int t = 0; t < 4; ++t) {
      c[t]     = wmma16(a0L, b0[t], c[t]);
      c[4 + t] = wmma16(a0H, b0[t], c[4 + t]);
    }
    if (kb + 64 < DMODEL)
      load_step(X, W, mbase, nbase, kb + 64, col, a0L, a0H, b0);
#pragma unroll
    for (int t = 0; t < 4; ++t) {
      c[t]     = wmma16(a1L, b1[t], c[t]);
      c[4 + t] = wmma16(a1H, b1[t], c[4 + t]);
    }
  }
}

// ---------------------------------------------------------------------------
// Stage 1: fused QKV projection.  y = x @ W^T + b, stored [B,H,S,64] f16.
// grid: 256 Mtiles * 12 Ntiles * 3 weights = 9216 waves = 1152 blocks.
// ---------------------------------------------------------------------------
__global__ __launch_bounds__(256) void qkv_gemm(
    const _Float16* __restrict__ xh,
    const _Float16* __restrict__ wq, const _Float16* __restrict__ wk,
    const _Float16* __restrict__ wv,
    const float* __restrict__ bq, const float* __restrict__ bk,
    const float* __restrict__ bv,
    _Float16* __restrict__ qh, _Float16* __restrict__ kh,
    _Float16* __restrict__ vh) {
  int wave = blockIdx.x * 8 + (threadIdx.x >> 5);
  int lane = threadIdx.x & 31;
  int col = lane & 15, g = lane >> 4;

  int nt  = wave % (DMODEL / 64);
  int mt  = (wave / (DMODEL / 64)) % (MM / 32);
  int sel = wave / ((DMODEL / 64) * (MM / 32));   // 0=Q 1=K 2=V (wave-uniform)

  const _Float16* W    = (sel == 0) ? wq : (sel == 1) ? wk : wv;
  const float*    bias = (sel == 0) ? bq : (sel == 1) ? bk : bv;
  _Float16*       outp = (sel == 0) ? qh : (sel == 1) ? kh : vh;

  int mbase = mt * 32, nbase = nt * 64;
  v8f c[8] = {};
  gemm_tile(xh, W, mbase, nbase, col, c);

#pragma unroll
  for (int t = 0; t < 4; ++t) {
    int n = nbase + 16 * t + col;
    float bn = bias[n];
    int h = n >> 6, d = n & 63;     // nbase is 64-aligned -> one head per wave
#pragma unroll
    for (int half = 0; half < 2; ++half) {
#pragma unroll
      for (int i = 0; i < 8; ++i) {
        int tok = mbase + 16 * half + i + 8 * g;
        int b_ = tok >> 12, s = tok & (SS - 1);
        float v = c[4 * half + t][i] + bn;
        outp[(((size_t)b_ * NH + h) * SS + s) * HD + d] = (_Float16)v;
      }
    }
  }
}

// ---------------------------------------------------------------------------
// Stage 2: causal flash attention.  One wave = 32 query rows of one (b,h).
// Per 32-key block: 8 WMMAs Q.K^T + 8 WMMAs P.V, one V staging + LDS sync.
// Next-block K fragments issued right after the V staging (overlap softmax/LDS).
// grid: 2*12*128 = 3072 waves = 384 blocks.  LDS: 3840 halfs/wave (60KB/WG).
// ---------------------------------------------------------------------------
__global__ __launch_bounds__(256) void attn_fwd(
    const _Float16* __restrict__ qh, const _Float16* __restrict__ kh,
    const _Float16* __restrict__ vh, _Float16* __restrict__ ph) {
  __shared__ __align__(16) _Float16 lds[8 * 3840];

  int wave = blockIdx.x * 8 + (threadIdx.x >> 5);
  int lane = threadIdx.x & 31;
  int col = lane & 15, g = lane >> 4;

  int qt = wave & (SS / 32 - 1);     // 128 q-tiles per (b,h)
  int bh = wave >> 7;
  int h = bh % NH, b = bh / NH;
  int qbase = qt * 32;

  const size_t head = ((size_t)(b * NH + h)) * SS * HD;
  const _Float16* Qp = qh + head;
  const _Float16* Kp = kh + head;
  const _Float16* Vp = vh + head;

  _Float16* Vt = &lds[(threadIdx.x >> 5) * 3840];  // [64][40]: V transposed
  _Float16* Pl = Vt + 64 * 40;                     // [32][40]: P tile

  // Q fragments: [row half r][hd chunk c], reused for every KV block
  v16h aq[2][2];
#pragma unroll
  for (int r = 0; r < 2; ++r)
#pragma unroll
    for (int cc = 0; cc < 2; ++cc)
      aq[r][cc] = ldfrag_g(Qp, HD, qbase + 16 * r + col, 32 * cc);

  v8f o[2][4] = {};
  float m[2][8], l[2][8];
#pragma unroll
  for (int r = 0; r < 2; ++r)
#pragma unroll
    for (int i = 0; i < 8; ++i) { m[r][i] = -__builtin_inff(); l[r][i] = 0.f; }

  const float lam = 0.18033688011112042f;  // log2(e) / sqrt(64)
  int nblk = (qbase >> 5) + 1;             // skip fully-masked KV blocks

  // pipelined K fragments for block 0: kf[2*t + c] (t = key col tile)
  v16h kf[4];
#pragma unroll
  for (int t = 0; t < 2; ++t)
#pragma unroll
    for (int cc = 0; cc < 2; ++cc)
      kf[2 * t + cc] = ldfrag_g(Kp, HD, 16 * t + col, 32 * cc);

  for (int jb = 0; jb < nblk; ++jb) {
    int jbase = jb * 32;

    // ---- S = Q . K^T : 32x32 logits (2 row halves x 2 col tiles) ----
    v8f c[2][2] = {};
#pragma unroll
    for (int r = 0; r < 2; ++r)
#pragma unroll
      for (int t = 0; t < 2; ++t) {
        c[r][t] = wmma16(aq[r][0], kf[2 * t + 0], c[r][t]);
        c[r][t] = wmma16(aq[r][1], kf[2 * t + 1], c[r][t]);
      }

    // ---- stage V block transposed into LDS: Vt[d][k], stride 40 ----
#pragma unroll
    for (int it = 0; it < 8; ++it) {
      int chunk = it * 32 + lane;          // 0..255 (k, 8-wide d chunk)
      int k  = chunk >> 3;
      int d0 = (chunk & 7) * 8;
      v8h pv = *(const v8h*)(Vp + (size_t)(jbase + k) * HD + d0);
#pragma unroll
      for (int e = 0; e < 8; ++e) Vt[(d0 + e) * 40 + k] = pv[e];
    }

    // ---- issue next block's K fragment loads (overlap softmax below) ----
    if (jb + 1 < nblk) {
      int jn = jbase + 32;
#pragma unroll
      for (int t = 0; t < 2; ++t)
#pragma unroll
        for (int cc = 0; cc < 2; ++cc)
          kf[2 * t + cc] = ldfrag_g(Kp, HD, jn + 16 * t + col, 32 * cc);
      __builtin_prefetch(Vp + (size_t)jn * HD, 0, 1);
    }

    // ---- causal mask + online softmax (C-fragment layout) ----
    bool need_mask = (jbase + 31) > qbase;   // only the diagonal block
#pragma unroll
    for (int r = 0; r < 2; ++r) {
      float alpha[8];
#pragma unroll
      for (int i = 0; i < 8; ++i) {
        int q = qbase + 16 * r + i + 8 * g;
        float s0 = c[r][0][i], s1 = c[r][1][i];
        if (need_mask) {
          s0 = (jbase + col > q)      ? -__builtin_inff() : s0;
          s1 = (jbase + 16 + col > q) ? -__builtin_inff() : s1;
        }
        float bm = rmax16(fmaxf(s0, s1));
        float mn = fmaxf(m[r][i], bm);
        alpha[i] = __builtin_exp2f((m[r][i] - mn) * lam);
        m[r][i] = mn;
        float p0 = __builtin_exp2f((s0 - mn) * lam);
        float p1 = __builtin_exp2f((s1 - mn) * lam);
        int row = 16 * r + i + 8 * g;
        Pl[row * 40 + col]      = (_Float16)p0;
        Pl[row * 40 + 16 + col] = (_Float16)p1;
        l[r][i] = l[r][i] * alpha[i] + rsum16(p0 + p1);
      }
#pragma unroll
      for (int t = 0; t < 4; ++t)
#pragma unroll
        for (int i = 0; i < 8; ++i) o[r][t][i] *= alpha[i];
    }

    asm volatile("s_wait_dscnt 0x0" ::: "memory");  // LDS stores visible

    // ---- O += P . V (contraction over 32 keys) ----
    {
      int sel = (lane < 16) ? 0 : 8;
      v16h ap0 = ldfrag(Pl + (size_t)col * 40 + sel);
      v16h ap1 = ldfrag(Pl + (size_t)(16 + col) * 40 + sel);
#pragma unroll
      for (int t = 0; t < 4; ++t) {
        v16h bv = ldfrag(Vt + (size_t)(16 * t + col) * 40 + sel);
        o[0][t] = wmma16(ap0, bv, o[0][t]);
        o[1][t] = wmma16(ap1, bv, o[1][t]);
      }
    }
  }

  // ---- normalize and store preout as f16 in [B,S,D] ----
#pragma unroll
  for (int r = 0; r < 2; ++r) {
#pragma unroll
    for (int i = 0; i < 8; ++i) l[r][i] = 1.0f / l[r][i];
#pragma unroll
    for (int t = 0; t < 4; ++t) {
      int d = h * HD + 16 * t + col;
#pragma unroll
      for (int i = 0; i < 8; ++i) {
        int tok = b * SS + qbase + 16 * r + i + 8 * g;
        ph[(size_t)tok * DMODEL + d] = (_Float16)(o[r][t][i] * l[r][i]);
      }
    }
  }
}

// ---------------------------------------------------------------------------
// Stage 3: output projection.  out = preout @ Wo^T + bo  (f32 output)
// grid: 256 * 12 = 3072 waves = 384 blocks.
// ---------------------------------------------------------------------------
__global__ __launch_bounds__(256) void out_gemm(
    const _Float16* __restrict__ ph, const _Float16* __restrict__ wo,
    const float* __restrict__ bo, float* __restrict__ out) {
  int wave = blockIdx.x * 8 + (threadIdx.x >> 5);
  int lane = threadIdx.x & 31;
  int col = lane & 15, g = lane >> 4;

  int nt = wave % (DMODEL / 64);
  int mt = wave / (DMODEL / 64);
  int mbase = mt * 32, nbase = nt * 64;

  v8f c[8] = {};
  gemm_tile(ph, wo, mbase, nbase, col, c);

#pragma unroll
  for (int t = 0; t < 4; ++t) {
    int n = nbase + 16 * t + col;
    float bn = bo[n];
#pragma unroll
    for (int half = 0; half < 2; ++half) {
#pragma unroll
      for (int i = 0; i < 8; ++i) {
        int tok = mbase + 16 * half + i + 8 * g;
        out[(size_t)tok * DMODEL + n] = c[4 * half + t][i] + bn;
      }
    }
  }
}

// ---------------------------------------------------------------------------
extern "C" void kernel_launch(void* const* d_in, const int* in_sizes, int n_in,
                              void* d_out, int out_size, void* d_ws,
                              size_t ws_size, hipStream_t stream) {
  const float* x  = (const float*)d_in[0];
  const float* Wq = (const float*)d_in[1];
  const float* bq = (const float*)d_in[2];
  const float* Wk = (const float*)d_in[3];
  const float* bk = (const float*)d_in[4];
  const float* Wv = (const float*)d_in[5];
  const float* bv = (const float*)d_in[6];
  const float* Wo = (const float*)d_in[7];
  const float* bo = (const float*)d_in[8];
  float* out = (float*)d_out;

  char* ws = (char*)d_ws;
  size_t off = 0;
  auto take = [&](size_t bytes) -> char* {
    char* p = ws + off;
    off += (bytes + 255) & ~(size_t)255;
    return p;
  };
  _Float16* xh  = (_Float16*)take((size_t)MM * DMODEL * 2);
  _Float16* wqh = (_Float16*)take((size_t)DMODEL * DMODEL * 2);
  _Float16* wkh = (_Float16*)take((size_t)DMODEL * DMODEL * 2);
  _Float16* wvh = (_Float16*)take((size_t)DMODEL * DMODEL * 2);
  _Float16* woh = (_Float16*)take((size_t)DMODEL * DMODEL * 2);
  _Float16* qh  = (_Float16*)take((size_t)MM * DMODEL * 2);
  _Float16* kh  = (_Float16*)take((size_t)MM * DMODEL * 2);
  _Float16* vh  = (_Float16*)take((size_t)MM * DMODEL * 2);
  _Float16* ph  = (_Float16*)take((size_t)MM * DMODEL * 2);

  cvt_f32_f16<<<1024, 256, 0, stream>>>(x, xh, MM * DMODEL);
  cvt_f32_f16<<<512, 256, 0, stream>>>(Wq, wqh, DMODEL * DMODEL);
  cvt_f32_f16<<<512, 256, 0, stream>>>(Wk, wkh, DMODEL * DMODEL);
  cvt_f32_f16<<<512, 256, 0, stream>>>(Wv, wvh, DMODEL * DMODEL);
  cvt_f32_f16<<<512, 256, 0, stream>>>(Wo, woh, DMODEL * DMODEL);

  // 256 Mtiles * 12 Ntiles * 3 weights / 8 waves = 1152 blocks
  qkv_gemm<<<1152, 256, 0, stream>>>(xh, wqh, wkh, wvh, bq, bk, bv, qh, kh, vh);
  // 2*12*128 waves / 8 = 384 blocks
  attn_fwd<<<384, 256, 0, stream>>>(qh, kh, vh, ph);
  // 256*12 waves / 8 = 384 blocks
  out_gemm<<<384, 256, 0, stream>>>(ph, woh, bo, out);
}
